// SubAttention_15625091022858
// MI455X (gfx1250) — matrix operand order repaired and checked by reading, not compiled, not run on hardware
//
#include <hip/hip_runtime.h>
#include <hip/hip_bf16.h>

typedef __attribute__((ext_vector_type(16))) __bf16 v16bf;
typedef __attribute__((ext_vector_type(8)))  float  v8f;

#define NP   16384   // num_p
#define NHD  9       // neighbors
#define HD   16      // head dim
#define PB   16      // positions per wave (= WMMA M)
#define HROW 68      // padded LDS row stride (bf16) for 64-wide hidden

static __device__ __forceinline__ v8f wmma_bf16(v16bf a, v16bf b, v8f c) {
    return __builtin_amdgcn_wmma_f32_16x16x32_bf16(
        /*neg_a=*/false, a, /*neg_b=*/false, b,
        /*c_mod=*/(short)0, c, /*reuse_a=*/false, /*reuse_b=*/false);
}

// Single-instruction ReLU / max via v_med3_f32 (avoids fmaxf's canonicalize+max).
static __device__ __forceinline__ float relu1(float x) {
    return __builtin_amdgcn_fmed3f(x, 0.0f, __builtin_inff());
}
static __device__ __forceinline__ float max1(float a, float b) {
    return __builtin_amdgcn_fmed3f(a, b, __builtin_inff());
}

__global__ __launch_bounds__(256) void subattn_wmma_kernel(
    const float* __restrict__ q,  const float* __restrict__ k,
    const float* __restrict__ v,  const float* __restrict__ pos,
    const float* __restrict__ W1, const float* __restrict__ b1,
    const float* __restrict__ W2, const float* __restrict__ b2,
    float* __restrict__ out)
{
    // per-wave LDS scratch: 16x64 bf16 hidden tile (row-padded to 68)
    __shared__ __bf16 hiddenS[8][PB * HROW];

    const int lane = threadIdx.x & 31;
    const int wave = threadIdx.x >> 5;
    __bf16* hS = &hiddenS[wave][0];

    // task = (b, h, pblk): 4 * 4 * 1024 = 16384 waves total
    const int task = blockIdx.x * 8 + wave;
    const int pblk = task & 1023;
    const int h    = (task >> 10) & 3;
    const int b    = task >> 12;
    const int p0   = pblk * PB;

    const int lh = lane >> 4;   // lane half
    const int lm = lane & 15;

    // ---- W1~ (17x64: W1 rows + b1 as row K=16) as B operand, 4 column blocks ----
    // B layout (32x16 bf16): lanes 0-15: elems 0..7 <- K 0..7, 8..15 <- K 16..23, N=lane;
    //                        lanes 16-31: elems 0..7 <- K 8..15, 8..15 <- K 24..31, N=lane-16.
    // K=16 row carries b1 (bias-row trick); K=17..31 are zero pad.
    v16bf w1b[4];
    #pragma unroll
    for (int nb = 0; nb < 4; ++nb) {
        v16bf t;
        #pragma unroll
        for (int e = 0; e < 16; ++e) {
            float val;
            if (e < 8)                 val = W1[(lh * 8 + e) * 64 + nb * 16 + lm];  // K = lh*8+e
            else if (e == 8 && lh == 0) val = b1[nb * 16 + lm];           // K = 16 -> bias
            else                       val = 0.0f;                        // K = 17..31 pad
            t[e] = (__bf16)val;
        }
        w1b[nb] = t;
    }
    // ---- W2^T (16x64: M=c, K=j) as A operand, two 16x32 K-chunks ----
    // A layout (16x32 bf16): lanes 0-15: M=lane, elems 0..7 <- K 0..7, 8..15 <- K 16..23
    //                        lanes 16-31: M=lane-16, elems 0..7 <- K 8..15, 8..15 <- K 24..31
    // b2 dropped: constant across neighbors -> cancels in softmax (exact).
    // W2 pre-scaled by log2(e): logits come out in log2 domain so softmax can use
    // raw v_exp_f32 (base-2) with no per-element multiply.
    const float LOG2E = 1.4426950408889634f;
    v16bf w2a[2];
    #pragma unroll
    for (int ck = 0; ck < 2; ++ck) {
        v16bf t;
        #pragma unroll
        for (int e = 0; e < 16; ++e) {
            int jo = (e & 7) + ((e >> 3) * 16) + lh * 8;  // K offset within chunk
            int j  = ck * 32 + jo;
            t[e] = (__bf16)(W2[j * 16 + lm] * LOG2E);     // W2^T[c=lm][j] * log2(e)
        }
        w2a[ck] = t;
    }
    (void)b2;

    // ---- per-lane row addressing: row M = lm, channels c0..c0+7 (c0 = lh*8) ----
    // Same pattern serves the x-stage (q-k+pos) AND the output stage (v+pos).
    const int c0 = lh * 8;
    const int rowBase = ((b * NP + p0 + lm) * NHD) * 64 + h * HD + c0;
    const int qbase   = (b * NP + p0 + lm) * 64 + h * HD + c0;
    float qv[8];
    #pragma unroll
    for (int e = 0; e < 8; ++e) qv[e] = q[qbase + e];

    // Constant-1 element feeding the bias row: lanes 0-15, element 8 <-> K=16.
    const __bf16 oneK16 = (__bf16)((lh == 0) ? 1.0f : 0.0f);
    const __bf16 zbf    = (__bf16)0.0f;

    v8f logits[NHD];   // D' layout: lane = p_local, VGPR g = channel (g + lh*8)

    #pragma unroll
    for (int n = 0; n < NHD; ++n) {
        // x~ = [q - k + pos, 1] : one 16-row tile for neighbor n (A operand of MLP1)
        const int kb = rowBase + n * 64;
        v16bf a;
        #pragma unroll
        for (int e = 0; e < 8; ++e) {
            float xv = qv[e] - k[kb + e] + pos[kb + e];
            a[e]     = (__bf16)xv;
            a[e + 8] = zbf;
        }
        a[8] = oneK16;   // K=16 -> multiplies the b1 row of W1~

        // MLP1: hidden(16x64) = relu(x~ @ W1~), four N-blocks, C = inline 0
        v8f hacc[4];
        #pragma unroll
        for (int nb = 0; nb < 4; ++nb) {
            v8f zc = {};
            hacc[nb] = wmma_bf16(a, w1b[nb], zc);
        }

        // ReLU (v_med3) + D-layout (lane=col j, VGPR=row p) -> LDS hidden[p][j]
        #pragma unroll
        for (int nb = 0; nb < 4; ++nb) {
            #pragma unroll
            for (int g = 0; g < 8; ++g) {
                float f = relu1(hacc[nb][g]);
                hS[(g + lh * 8) * HROW + nb * 16 + lm] = (__bf16)f;
            }
        }
        // Read back as B operand hidden^T (64x16): lane = p_local, j in table order
        // (elems 0..7 <- j lh*8.., 8..15 <- j 16+lh*8..) per K-chunk. Same-wave DS ops are in-order.
        v16bf hB0, hB1;
        #pragma unroll
        for (int e = 0; e < 16; ++e) {
            hB0[e] = hS[lm * HROW +      (e & 7) + (e >> 3) * 16 + lh * 8];   // K = j 0..31
            hB1[e] = hS[lm * HROW + 32 + (e & 7) + (e >> 3) * 16 + lh * 8];   // K = j 32..63
        }

        // MLP2 (swapped): log2-logits^T(16x16) = (W2^T*log2e) @ hidden^T, C = inline 0
        v8f zc = {};
        v8f l = wmma_bf16(w2a[0], hB0, zc);
        l = wmma_bf16(w2a[1], hB1, l);
        logits[n] = l;
    }

    // ---- softmax over 9 neighbors (log2 domain): elementwise in registers ----
    v8f mx = logits[0];
    #pragma unroll
    for (int n = 1; n < NHD; ++n)
        #pragma unroll
        for (int g = 0; g < 8; ++g) mx[g] = max1(mx[g], logits[n][g]);
    float ssum[8];
    #pragma unroll
    for (int g = 0; g < 8; ++g) ssum[g] = 0.0f;
    #pragma unroll
    for (int n = 0; n < NHD; ++n)
        #pragma unroll
        for (int g = 0; g < 8; ++g) {
            float e = __builtin_amdgcn_exp2f(logits[n][g] - mx[g]);  // v_exp_f32
            logits[n][g] = e;               // unnormalized weights
            ssum[g] += e;
        }
    #pragma unroll
    for (int g = 0; g < 8; ++g) ssum[g] = __builtin_amdgcn_rcpf(ssum[g]);  // v_rcp_f32

    // ---- out[row lm][c0..c0+7] = sum_n attn[n] * (v[n] + pos[n]) ----
    // Contiguous 8-float loads per neighbor (b128), pos re-read hits L2.
    float acc[8];
    #pragma unroll
    for (int g = 0; g < 8; ++g) acc[g] = 0.0f;
    #pragma unroll
    for (int n = 0; n < NHD; ++n) {
        const int idx = rowBase + n * 64;
        #pragma unroll
        for (int g = 0; g < 8; ++g)
            acc[g] += logits[n][g] * (v[idx + g] + pos[idx + g]);
    }
    #pragma unroll
    for (int g = 0; g < 8; ++g)
        out[qbase + g] = acc[g] * ssum[g];   // contiguous b128 stores
}

extern "C" void kernel_launch(void* const* d_in, const int* in_sizes, int n_in,
                              void* d_out, int out_size, void* d_ws, size_t ws_size,
                              hipStream_t stream) {
    const float* q   = (const float*)d_in[0];
    const float* k   = (const float*)d_in[1];
    const float* v   = (const float*)d_in[2];
    const float* pos = (const float*)d_in[3];
    const float* W1  = (const float*)d_in[4];
    const float* b1  = (const float*)d_in[5];
    const float* W2  = (const float*)d_in[6];
    const float* b2  = (const float*)d_in[7];
    float* out = (float*)d_out;

    // 4 batches * 4 heads * 1024 position-blocks = 16384 waves; 8 waves/block
    dim3 grid(2048), block(256);
    hipLaunchKernelGGL(subattn_wmma_kernel, grid, block, 0, stream,
                       q, k, v, pos, W1, b1, W2, b2, out);
}
